// RNNPolicy_22239340659035
// MI455X (gfx1250) — matrix operand order repaired and checked by reading, not compiled, not run on hardware
//
#include <hip/hip_runtime.h>
#include <hip/hip_bf16.h>
#include <math.h>

#define HID    20
#define FEAT   3
#define BATCH  1024
#define TSTEPS 4096
#define KDIM   32          // padded K: h = 0..19, x = 20..22, zeros = 23..31
#define NGATE  80          // 4*HID, torch gate order i,f,g,o
#define NTILE  5           // 80 / 16 N-tiles
#define MT     16          // assets per wave tile

typedef __attribute__((ext_vector_type(16))) _Float16 v16h;
typedef __attribute__((ext_vector_type(8)))  _Float16 v8h;
typedef __attribute__((ext_vector_type(2)))  _Float16 v2h;
typedef __attribute__((ext_vector_type(8)))  float    v8f;
typedef __attribute__((ext_vector_type(4)))  float    v4f;

// fast sigmoid/tanh: v_exp_f32 + v_rcp_f32, no IEEE div expansion
__device__ __forceinline__ float sigf(float x) {
    return __builtin_amdgcn_rcpf(1.0f + __expf(-x));
}
__device__ __forceinline__ float tanhfast(float x) {
    return 1.0f - 2.0f * __builtin_amdgcn_rcpf(__expf(2.0f * x) + 1.0f);
}

// One wave (32 threads) per block; each block owns 16 assets and runs the full
// 4096-step recurrence with 5 v_wmma_f32_16x16x32_f16 per step.
__global__ __launch_bounds__(32) void lstm_wmma_kernel(
    const float* __restrict__ x,      // (FEAT, B, T)
    const float* __restrict__ W_ih,   // (80, 3)
    const float* __restrict__ W_hh,   // (80, 20)
    const float* __restrict__ b_ih,   // (80,)
    const float* __restrict__ b_hh,   // (80,)
    float* __restrict__ h_out)        // (B, HID) f32
{
    __shared__ __align__(16) _Float16 Wc[KDIM][NGATE];    // combined [k][n] weights (5 KB)
    __shared__ __align__(16) _Float16 hbuf[MT][KDIM];     // A rows: [h | x_t | 0] as f16 (1 KB)
    __shared__ __align__(16) float    gatesT[NGATE][MT];  // gates transposed [n][m] (5 KB)

    const int  lane   = threadIdx.x;       // wave32
    const int  tile   = blockIdx.x;        // 0..63
    const int  mlo    = lane & 15;
    const bool hiHalf = lane >= 16;
    const int  basset = tile * MT;

    // ---- one-time: combined weight Wc[k][n] = (k<20 ? W_hh[n][k] : W_ih[n][k-20]), pad 0 ----
    for (int idx = lane; idx < KDIM * NGATE; idx += 32) {
        int k = idx / NGATE, n = idx % NGATE;
        float w = 0.0f;
        if (k < HID)             w = W_hh[n * HID + k];
        else if (k < HID + FEAT) w = W_ih[n * FEAT + (k - HID)];
        Wc[k][n] = (_Float16)w;
    }
    for (int idx = lane; idx < MT * KDIM; idx += 32)
        ((_Float16*)hbuf)[idx] = (_Float16)0.0f;
    __syncthreads();

    // ---- persistent per-lane B fragments (K-major gfx1250 layout):
    // lanes 0-15: N=lane, elems 0..7 = K0..7, 8..15 = K16..23; lanes 16-31: N=lane-16, elems 0..7 = K8..15, 8..15 = K24..31
    v16h bfrag[NTILE];
    {
        const int kbase = hiHalf ? 8 : 0;
        for (int j = 0; j < NTILE; ++j) {
            const int n = j * 16 + mlo;
            v16h b;
            for (int e = 0; e < 16; ++e) b[e] = Wc[(e < 8 ? kbase + e : 16 + kbase + (e - 8))][n];
            bfrag[j] = b;
        }
    }
    // ---- bias folded into C accumulator: C/D lane holds a single N column ----
    float biasv[NTILE];
    for (int j = 0; j < NTILE; ++j) {
        const int n = j * 16 + mlo;
        biasv[j] = b_ih[n] + b_hh[n];
    }

    // ---- per-lane state: lane owns 10 consecutive cells of one asset row ----
    // p = lane*10+q  =>  m = lane>>1, cell = (lane&1)*10 + q  (q = 0..9, same row)
    const int em     = lane >> 1;
    const int ecell0 = (lane & 1) * 10;
    float* __restrict__    gbase  = &gatesT[ecell0][em];   // + (q + g*HID)*MT, imm offsets
    _Float16* __restrict__ hstore = &hbuf[em][ecell0];     // + q, imm offsets

    float creg[10], hreg[10];
#pragma unroll
    for (int q = 0; q < 10; ++q) { creg[q] = 0.0f; hreg[q] = 0.0f; }

    // A-fragment LDS pointers (loop-invariant), 16-bit A 16x32 layout:
    // lanes 0-15: row=lane,    elems 0..7 = K0..7,  8..15 = K16..23
    // lanes16-31: row=lane-16, elems 0..7 = K8..15, 8..15 = K24..31
    const v8h* __restrict__ apA = (const v8h*)&hbuf[mlo][hiHalf ? 8 : 0];
    const v8h* __restrict__ apB = (const v8h*)&hbuf[mlo][hiHalf ? 24 : 16];

    // x staging: slot 22 gets x[2]; lanes >=16 write an f16 zero into padding slot 24
    // (padding must stay zero) so all 32 lanes run identical straight-line code.
    _Float16* __restrict__ xs01 = &hbuf[mlo][HID + (lane >> 4)];
    _Float16* __restrict__ xs2  = &hbuf[mlo][hiHalf ? 24 : 22];

    const long long BT  = (long long)BATCH * TSTEPS;
    const float* __restrict__ xr0 = x + (long long)(lane >> 4) * BT + (long long)(basset + mlo) * TSTEPS;
    const float* __restrict__ xr2 = x + 2ll * BT + (long long)(basset + mlo) * TSTEPS;

    // software pipeline: x for step t is loaded during step t-1
    float xcur0 = xr0[0];
    float xcur2 = xr2[0];

    for (int t = 0; t < TSTEPS; ++t) {
        // stage x_t into A rows (K = 20..22)
        *xs01 = (_Float16)xcur0;
        *xs2  = hiHalf ? (_Float16)0.0f : (_Float16)xcur2;

        // issue next step's loads now; consumed only at next iteration's stores,
        // so L2 latency hides behind this step's WMMA + elementwise work
        const int tn = (t + 1 < TSTEPS) ? (t + 1) : (TSTEPS - 1);
        const float xn0 = xr0[tn];
        const float xn2 = xr2[tn];

        __builtin_amdgcn_wave_barrier();   // single wave: LDS in-order; pin program order

        union { v16h v; v8h h[2]; } af;
        af.h[0] = *apA;
        af.h[1] = *apB;

        // gates = [h|x] @ [W_hh.T ; W_ih.T] + bias via 5 WMMAs (N = 0..79)
#pragma unroll
        for (int j = 0; j < NTILE; ++j) {
            const float bv = biasv[j];
            v8f c = {bv, bv, bv, bv, bv, bv, bv, bv};
            v8f d = __builtin_amdgcn_wmma_f32_16x16x32_f16(
                false, af.v, false, bfrag[j], (short)0, c, false, false);
            // D layout: VGPR r -> M = r (lanes 0-15) or 8+r (lanes 16-31), N = lane&15
            const int n  = j * 16 + mlo;
            const int m0 = hiHalf ? 8 : 0;
            v4f lo = {d[0], d[1], d[2], d[3]};
            v4f hi = {d[4], d[5], d[6], d[7]};
            *(v4f*)&gatesT[n][m0]     = lo;
            *(v4f*)&gatesT[n][m0 + 4] = hi;
        }
        __builtin_amdgcn_wave_barrier();

        // elementwise cell update: fully unrolled, base + immediate LDS offsets
        _Float16 hh16[10];
#pragma unroll
        for (int q = 0; q < 10; ++q) {
            const float gi = gbase[q * MT];
            const float gf = gbase[(q + HID) * MT];
            const float gg = gbase[(q + 2 * HID) * MT];
            const float go = gbase[(q + 3 * HID) * MT];
            const float cc = sigf(gf) * creg[q] + sigf(gi) * tanhfast(gg);
            const float hh = sigf(go) * tanhfast(cc);
            creg[q] = cc;
            hreg[q] = hh;
            hh16[q] = (_Float16)hh;
        }
        // 5 packed b32 stores instead of 10 b16 stores (4B aligned)
#pragma unroll
        for (int q = 0; q < 10; q += 2) {
            v2h p2 = {hh16[q], hh16[q + 1]};
            *(v2h*)&hstore[q] = p2;
        }
        __builtin_amdgcn_wave_barrier();

        xcur0 = xn0;
        xcur2 = xn2;
    }

    // final hidden state, f32, to workspace
#pragma unroll
    for (int q = 0; q < 10; ++q)
        h_out[(basset + em) * HID + ecell0 + q] = hreg[q];
}

// Head: y = relu(h) @ fc_w.T + fc_b ; z = [lastAction, y] . conv_w + conv_b ;
// softmax over [1.0, z_0..z_1023]. One block of 1024 threads (thread == asset).
__global__ __launch_bounds__(1024) void policy_head_kernel(
    const float* __restrict__ h_last,      // (B, HID)
    const float* __restrict__ lastAction,  // (B,)
    const float* __restrict__ fc_w,        // (HID, HID)
    const float* __restrict__ fc_b,        // (HID,)
    const float* __restrict__ conv_w,      // (HID+1,)
    const float* __restrict__ conv_b,      // (1,)
    float* __restrict__ out)               // (B+1,)
{
    __shared__ float red[BATCH];
    __shared__ float swc[HID + 1];
    __shared__ float sfw[HID * HID];
    __shared__ float sfb[HID];
    const int tid = threadIdx.x;

    if (tid < HID * HID) sfw[tid] = fc_w[tid];
    if (tid < HID)       sfb[tid] = fc_b[tid];
    if (tid < HID + 1)   swc[tid] = conv_w[tid];
    __syncthreads();

    float h[HID];
    for (int k = 0; k < HID; ++k) {
        const float v = h_last[tid * HID + k];
        h[k] = v > 0.0f ? v : 0.0f;
    }
    float z = lastAction[tid] * swc[0] + conv_b[0];
    for (int j = 0; j < HID; ++j) {
        float y = sfb[j];
        for (int k = 0; k < HID; ++k) y += h[k] * sfw[j * HID + k];
        z += y * swc[j + 1];
    }

    // block-wide softmax over 1025 logits (logit 0 == 1.0 cash asset)
    red[tid] = z;
    __syncthreads();
    for (int s = 512; s > 0; s >>= 1) {
        if (tid < s) red[tid] = fmaxf(red[tid], red[tid + s]);
        __syncthreads();
    }
    const float mx = fmaxf(red[0], 1.0f);
    __syncthreads();
    const float ez = __expf(z - mx);
    red[tid] = ez;
    __syncthreads();
    for (int s = 512; s > 0; s >>= 1) {
        if (tid < s) red[tid] += red[tid + s];
        __syncthreads();
    }
    const float e0  = __expf(1.0f - mx);
    const float inv = 1.0f / (red[0] + e0);
    out[tid + 1] = ez * inv;
    if (tid == 0) out[0] = e0 * inv;
}

extern "C" void kernel_launch(void* const* d_in, const int* in_sizes, int n_in,
                              void* d_out, int out_size, void* d_ws, size_t ws_size,
                              hipStream_t stream) {
    const float* x   = (const float*)d_in[0];
    const float* la  = (const float*)d_in[1];
    const float* Wih = (const float*)d_in[2];
    const float* Whh = (const float*)d_in[3];
    const float* bih = (const float*)d_in[4];
    const float* bhh = (const float*)d_in[5];
    const float* fcw = (const float*)d_in[6];
    const float* fcb = (const float*)d_in[7];
    const float* cvw = (const float*)d_in[8];
    const float* cvb = (const float*)d_in[9];
    float* out = (float*)d_out;
    float* hws = (float*)d_ws;   // B*HID f32 = 80 KB scratch for h_last

    lstm_wmma_kernel<<<BATCH / MT, 32, 0, stream>>>(x, Wih, Whh, bih, bhh, hws);
    policy_head_kernel<<<1, BATCH, 0, stream>>>(hws, la, fcw, fcb, cvw, cvb, out);
}
